// CRF_23819888623936
// MI455X (gfx1250) — matrix operand order repaired and checked by reading, not compiled, not run on hardware
//
#include <hip/hip_runtime.h>
#include <math.h>

// ---------------------------------------------------------------------------
// CRF loss for B=512, T=512, C=96 on MI455X (gfx1250, wave32, WMMA).
//
// log-partition via scaled forward algorithm, carried in BASE-2 so the raw
// hardware transcendentals (v_log_f32 = log2, v_exp_f32 = exp2) are used
// without IEEE range-fixup code:
//   precompute mtr[j] = max_i trans[i][j],  E[i][j] = exp(trans[i][j]-mtr[j])
//   state: p[b][i] = 2^(v2[b][i] - ofs2[b]),  max_i p = 1
//   step:  S[b][j]  = sum_i p[b][i]*E[i][j]        <-- 16x96x96 f16 WMMA GEMM
//          u2[b][j] = mtr[j]*log2e + log2(S) + emit[b][t][j]*log2e
//          m2 = max_j u2 ; p' = exp2(u2-m2) ; ofs2 += m2
//   logden = ln2 * (ofs2 + log2(sum_j p[j]*exp(end[j])))
// One workgroup per 16 batches (32 WGs), 6 waves, each wave owns 16 columns:
// 3 x v_wmma_f32_16x16x32_f16 per step, constant B fragments held in VGPRs.
// Cross-lane row reductions use DPP16 (pure VALU, no LDS-pipe bpermute).
// ---------------------------------------------------------------------------

#define B_ 512
#define T_ 512
#define C_ 96
#define TILE_M 16
#define ROWPAD 104            // 96 + 8 halves: 208B rows -> conflict-free b128
#define FWD_THREADS 192       // 6 waves of 32

#define LOG2E 1.4426950408889634f
#define LN2   0.6931471805599453f

typedef __attribute__((ext_vector_type(16))) _Float16 v16h;
typedef __attribute__((ext_vector_type(8)))  float    v8f;

// 16-lane (DPP row) tree reductions: quad_perm xor1, quad_perm xor2,
// row_half_mirror (8-swap), row_mirror (16-swap). Valid because after each
// stage every lane holds the reduction of its subgroup.
__device__ __forceinline__ float dpp_max16(float x) {
  float t;
  t = __int_as_float(__builtin_amdgcn_mov_dpp(__float_as_int(x), 0xB1, 0xF, 0xF, true));
  x = fmaxf(x, t);
  t = __int_as_float(__builtin_amdgcn_mov_dpp(__float_as_int(x), 0x4E, 0xF, 0xF, true));
  x = fmaxf(x, t);
  t = __int_as_float(__builtin_amdgcn_mov_dpp(__float_as_int(x), 0x141, 0xF, 0xF, true));
  x = fmaxf(x, t);
  t = __int_as_float(__builtin_amdgcn_mov_dpp(__float_as_int(x), 0x140, 0xF, 0xF, true));
  x = fmaxf(x, t);
  return x;
}

__device__ __forceinline__ float dpp_sum16(float x) {
  float t;
  t = __int_as_float(__builtin_amdgcn_mov_dpp(__float_as_int(x), 0xB1, 0xF, 0xF, true));
  x += t;
  t = __int_as_float(__builtin_amdgcn_mov_dpp(__float_as_int(x), 0x4E, 0xF, 0xF, true));
  x += t;
  t = __int_as_float(__builtin_amdgcn_mov_dpp(__float_as_int(x), 0x141, 0xF, 0xF, true));
  x += t;
  t = __int_as_float(__builtin_amdgcn_mov_dpp(__float_as_int(x), 0x140, 0xF, 0xF, true));
  x += t;
  return x;
}

// ---------------------------------------------------------------------------
// Precompute column maxima and exponentiated transition matrix (f16).
__global__ void crf_prep(const float* __restrict__ trans,
                         _Float16* __restrict__ E,
                         float* __restrict__ mtr) {
  int j = threadIdx.x;
  if (j >= C_) return;
  float m = -INFINITY;
  #pragma unroll 4
  for (int i = 0; i < C_; ++i) m = fmaxf(m, trans[i * C_ + j]);
  mtr[j] = m;
  #pragma unroll 4
  for (int i = 0; i < C_; ++i)
    E[i * C_ + j] = (_Float16)__expf(trans[i * C_ + j] - m);
}

// ---------------------------------------------------------------------------
// Forward scan (log partition) for one tile of 16 batches per workgroup.
__launch_bounds__(FWD_THREADS)
__global__ void crf_forward(const float* __restrict__ emissions,
                            const int* __restrict__ mask,
                            const _Float16* __restrict__ E,
                            const float* __restrict__ mtr,
                            const float* __restrict__ startT,
                            const float* __restrict__ endT,
                            float* __restrict__ logden) {
  __shared__ __align__(16) _Float16 A_lds[TILE_M * ROWPAD]; // p state, f16
  __shared__ float rowmax[2][TILE_M];                       // double-buffered
  __shared__ float rowsum[TILE_M];
  __shared__ float ofs[TILE_M];                             // base-2 offset

  const int tid  = threadIdx.x;
  const int wave = tid >> 5;
  const int l    = tid & 31;
  const int half = l >> 4;      // 0: lanes 0-15, 1: lanes 16-31
  const int nl   = l & 15;
  const int n    = wave * 16 + nl;           // this lane's output column
  const int b0   = blockIdx.x * TILE_M;

  if (tid < TILE_M) {
    rowmax[0][tid] = -INFINITY;
    rowmax[1][tid] = -INFINITY;
    rowsum[tid] = 0.0f;
    ofs[tid]    = 0.0f;
  }

  const float mtr2_n   = mtr[n] * LOG2E;     // base-2 column max of trans
  const float start_n  = startT[n];
  const float expend_n = __expf(endT[n]);

  // Constant B fragments (E is K-major 96x96). Dense 16-bit B 32x16 layout:
  // lane = N%16; lanes 0-15 hold K 0-7 & 16-23, lanes 16-31 hold K 8-15 & 24-31.
  v16h bfrag[3];
  #pragma unroll
  for (int f = 0; f < 3; ++f) {
    union { v16h v; _Float16 h[16]; } bu;
    const int kb = f * 32 + (half ? 8 : 0);
    #pragma unroll
    for (int h = 0; h < 16; ++h)
      bu.h[h] = E[(size_t)(kb + h + (h >= 8 ? 8 : 0)) * C_ + n];
    bfrag[f] = bu.v;
  }

  // This lane's 8 batch rows are r = half*8 + k (matches C/D VGPR layout).
  const float* emb = emissions + (size_t)(b0 + half * 8) * T_ * C_ + n;
  const int*   mkb = mask      + (size_t)(b0 + half * 8) * T_;
  const size_t rs  = (size_t)T_ * C_;

  float e_cur[8], e_nxt[8];
  int   mk_cur[8], mk_nxt[8];
  #pragma unroll
  for (int k = 0; k < 8; ++k) e_cur[k] = emb[k * rs];
  #pragma unroll
  for (int k = 0; k < 8; ++k) e_nxt[k] = emb[k * rs + C_];
  #pragma unroll
  for (int k = 0; k < 8; ++k) mk_nxt[k] = mkb[k * T_ + 1];

  __syncthreads();

  // ---- step 0: u2 = (start + emit[:,0]) * log2e ----
  {
    float u[8], m8[8];
    #pragma unroll
    for (int k = 0; k < 8; ++k) u[k] = (start_n + e_cur[k]) * LOG2E;
    #pragma unroll
    for (int k = 0; k < 8; ++k) m8[k] = dpp_max16(u[k]);
    if (nl == 0) {
      #pragma unroll
      for (int k = 0; k < 8; ++k)
        atomicMax(&rowmax[0][half * 8 + k], m8[k]);   // ds float max
    }
    __syncthreads();
    #pragma unroll
    for (int k = 0; k < 8; ++k) {
      const int row = half * 8 + k;
      A_lds[row * ROWPAD + n] =
          (_Float16)__builtin_amdgcn_exp2f(u[k] - rowmax[0][row]);
    }
    if (wave == 0 && nl == 0) {
      #pragma unroll
      for (int k = 0; k < 8; ++k) {
        const int row = half * 8 + k;
        ofs[row] += rowmax[0][row];
      }
    }
    __syncthreads();
  }

  // ---- steps 1..T-1: GEMM + renormalize ----
  for (int s = 1; s < T_; ++s) {
    #pragma unroll
    for (int k = 0; k < 8; ++k) { e_cur[k] = e_nxt[k]; mk_cur[k] = mk_nxt[k]; }
    if (s + 1 < T_) {               // software-pipeline next step's emissions
      #pragma unroll
      for (int k = 0; k < 8; ++k) e_nxt[k] = emb[k * rs + (size_t)(s + 1) * C_];
      #pragma unroll
      for (int k = 0; k < 8; ++k) mk_nxt[k] = mkb[k * T_ + (s + 1)];
    }

    // A fragments from LDS per ISA 16-bit A 16x32 layout:
    // lanes 0-15: K 0-7 & 16-23; lanes 16-31: K 8-15 & 24-31 (row M = nl).
    v8f acc = {};
    #pragma unroll
    for (int f = 0; f < 3; ++f) {
      union { v16h v; uint4 q[2]; } au;
      const int k1 = f * 32 + (half ? 8 : 0);
      const int k2 = f * 32 + (half ? 24 : 16);
      au.q[0] = *(const uint4*)&A_lds[nl * ROWPAD + k1];
      au.q[1] = *(const uint4*)&A_lds[nl * ROWPAD + k2];
      acc = __builtin_amdgcn_wmma_f32_16x16x32_f16(
          false, au.v, false, bfrag[f], (short)0, acc, false, false);
    }

    // S is in [~2^-9, 96]: no denorms/infinities -> raw v_log_f32 is safe.
    float u[8], m8[8];
    #pragma unroll
    for (int k = 0; k < 8; ++k)
      u[k] = mtr2_n + __builtin_amdgcn_logf(acc[k]) + e_cur[k] * LOG2E;

    #pragma unroll
    for (int k = 0; k < 8; ++k) m8[k] = dpp_max16(u[k]);

    const int buf = s & 1;
    if (nl == 0) {
      #pragma unroll
      for (int k = 0; k < 8; ++k)
        atomicMax(&rowmax[buf][half * 8 + k], m8[k]);
    }
    __syncthreads();  // barrier1: A reads of this step done, rowmax final

    #pragma unroll
    for (int k = 0; k < 8; ++k) {
      const int row = half * 8 + k;
      if (mk_cur[k])
        A_lds[row * ROWPAD + n] =
            (_Float16)__builtin_amdgcn_exp2f(u[k] - rowmax[buf][row]);
    }
    if (wave == 0 && nl == 0) {
      #pragma unroll
      for (int k = 0; k < 8; ++k) {
        const int row = half * 8 + k;
        if (mk_cur[k]) ofs[row] += rowmax[buf][row];
      }
    }
    if (tid < TILE_M) rowmax[buf ^ 1][tid] = -INFINITY;  // reset other buffer
    __syncthreads();  // barrier2: p' visible for next step's fragment loads
  }

  // ---- finalize: logden[b] = ln2*(ofs2 + log2( sum_j p[j]*exp(end[j]) )) --
  #pragma unroll
  for (int k = 0; k < 8; ++k) {
    const int row = half * 8 + k;
    float p = dpp_sum16((float)A_lds[row * ROWPAD + n] * expend_n);
    if (nl == 0) atomicAdd(&rowsum[row], p);
  }
  __syncthreads();
  if (tid < TILE_M)
    logden[b0 + tid] =
        LN2 * (ofs[tid] + __builtin_amdgcn_logf(rowsum[tid]));
}

// ---------------------------------------------------------------------------
// Numerator (joint likelihood): one block per batch, gather-reduce over T.
__global__ void crf_num(const float* __restrict__ emissions,
                        const long long* __restrict__ tags,
                        const int* __restrict__ mask,
                        const float* __restrict__ trans,
                        const float* __restrict__ startT,
                        const float* __restrict__ endT,
                        float* __restrict__ lognum) {
  const int b = blockIdx.x;
  const int tid = threadIdx.x;  // 128 threads
  __shared__ float red[128];
  __shared__ int   redm[128];
  const long long* tg = tags + (size_t)b * T_;
  const int* mk = mask + (size_t)b * T_;
  const float* em = emissions + (size_t)b * T_ * C_;

  float acc = 0.0f;
  int mcnt = 0;
  for (int t = tid; t < T_; t += 128) {
    const int mt = mk[t];
    mcnt += mt;
    if (t >= 1 && mt) {
      const int prev = (int)tg[t - 1];
      const int cur  = (int)tg[t];
      acc += trans[prev * C_ + cur] + em[(size_t)t * C_ + cur];
    }
  }
  red[tid] = acc; redm[tid] = mcnt;
  __syncthreads();
  for (int st = 64; st > 0; st >>= 1) {
    if (tid < st) { red[tid] += red[tid + st]; redm[tid] += redm[tid + st]; }
    __syncthreads();
  }
  if (tid == 0) {
    const int t0 = (int)tg[0];
    const int last = redm[0] - 1;
    const int lt = (int)tg[last];
    lognum[b] = red[0] + startT[t0] + em[t0] + endT[lt];
  }
}

// ---------------------------------------------------------------------------
// Final reduction: mean(logden - lognum) over 512 batches.
__global__ void crf_final(const float* __restrict__ logden,
                          const float* __restrict__ lognum,
                          float* __restrict__ out) {
  __shared__ float red[256];
  const int tid = threadIdx.x;
  float a = (logden[tid] - lognum[tid]) +
            (logden[tid + 256] - lognum[tid + 256]);
  red[tid] = a;
  __syncthreads();
  for (int st = 128; st > 0; st >>= 1) {
    if (tid < st) red[tid] += red[tid + st];
    __syncthreads();
  }
  if (tid == 0) out[0] = red[0] / (float)B_;
}

// ---------------------------------------------------------------------------
extern "C" void kernel_launch(void* const* d_in, const int* in_sizes, int n_in,
                              void* d_out, int out_size, void* d_ws, size_t ws_size,
                              hipStream_t stream) {
  (void)in_sizes; (void)n_in; (void)out_size; (void)ws_size;
  const float*     emissions = (const float*)d_in[0];
  const long long* tags      = (const long long*)d_in[1];
  const int*       mask      = (const int*)d_in[2];
  const float*     trans     = (const float*)d_in[3];
  const float*     startT    = (const float*)d_in[4];
  const float*     endT      = (const float*)d_in[5];
  float* out = (float*)d_out;

  // Workspace carve-up (23 KB total)
  char* ws = (char*)d_ws;
  _Float16* E   = (_Float16*)ws;              // 96*96*2  = 18432 B
  float* mtr    = (float*)(ws + 18432);       // 96*4     =   384 B
  float* logden = (float*)(ws + 18816);       // 512*4    =  2048 B
  float* lognum = (float*)(ws + 20864);       // 512*4    =  2048 B

  crf_prep   <<<1, 96, 0, stream>>>(trans, E, mtr);
  crf_forward<<<B_ / TILE_M, FWD_THREADS, 0, stream>>>(emissions, mask, E, mtr,
                                                       startT, endT, logden);
  crf_num    <<<B_, 128, 0, stream>>>(emissions, tags, mask, trans,
                                      startT, endT, lognum);
  crf_final  <<<1, 256, 0, stream>>>(logden, lognum, out);
}
